// FusedColorMLP_81827716923549
// MI455X (gfx1250) — matrix operand order repaired and checked by reading, not compiled, not run on hardware
//
#include <hip/hip_runtime.h>
#include <hip/hip_bf16.h>

typedef __attribute__((ext_vector_type(16))) _Float16 v16h;
typedef __attribute__((ext_vector_type(8)))  _Float16 v8h;
typedef __attribute__((ext_vector_type(2)))  _Float16 v2h;
typedef __attribute__((ext_vector_type(8)))  float    v8f;
typedef __attribute__((ext_vector_type(8)))  unsigned int v8u;

#define WAVES_PER_BLOCK 8
#define SWZ_SWAPX16 0x401F   // ds_swizzle group-of-32: xor=0x10, or=0, and=0x1F

// Per-wave DS ops are in-order on CDNA5; fence compiler scheduling + drain LDS
// between the enc store phase and the dependent B-fragment loads.
__device__ __forceinline__ void lds_phase_fence() {
    asm volatile("s_wait_dscnt 0" ::: "memory");
}

// Exchange a b32 between lane l and lane l^16 (half-wave swap).
__device__ __forceinline__ unsigned swapx16(unsigned v) {
    return (unsigned)__builtin_amdgcn_ds_swizzle((int)v, SWZ_SWAPX16);
}

// Two accumulator floats -> v_cvt_pk_f16_f32 -> ReLU via v_pk_max_num_f16.
__device__ __forceinline__ unsigned pack_relu(float a, float b) {
    v2h p; p[0] = (_Float16)a; p[1] = (_Float16)b;
    const v2h z = {};
    p = __builtin_elementwise_max(p, z);
    return __builtin_bit_cast(unsigned, p);
}

// A-fragment (16x32 f16) element->K mapping (contiguous, matches B):
// lanes0-15: K0..15 ; lanes16-31: K16..31
__device__ __forceinline__ int a_klocal(int e, int hi) {
    return hi * 16 + e;
}

// D (4 blocks, rows = features mb*16 + r + hi*8, lane = point) -> two B
// fragments of h^T (K=64 split as 2x32; B element e <-> k = hi*16+e), with
// ReLU. Pure register path: cvt_pk + pk_max + SWAPX16 swizzle + cndmask.
__device__ __forceinline__ void transpose_relu_to_bfrags(const v8f acc[4], int hi,
                                                         v16h bf[2]) {
    unsigned pk[4][4], sw[4][4];
#pragma unroll
    for (int mb = 0; mb < 4; ++mb)
#pragma unroll
        for (int rr = 0; rr < 4; ++rr)
            pk[mb][rr] = pack_relu(acc[mb][2 * rr], acc[mb][2 * rr + 1]);
#pragma unroll
    for (int mb = 0; mb < 4; ++mb)
#pragma unroll
        for (int rr = 0; rr < 4; ++rr)
            sw[mb][rr] = swapx16(pk[mb][rr]);
    const bool up = hi != 0;
#pragma unroll
    for (int kk = 0; kk < 2; ++kk) {
        v8u u;
#pragma unroll
        for (int j = 0; j < 4; ++j) {
            u[j]     = up ? sw[2 * kk + 1][j] : pk[2 * kk][j];
            u[4 + j] = up ? pk[2 * kk + 1][j] : sw[2 * kk][j];
        }
        bf[kk] = __builtin_bit_cast(v16h, u);
    }
}

__global__ __launch_bounds__(256) void
fused_color_mlp_kernel(const float* __restrict__ x,
                       const float* __restrict__ W1,   // (32,64) row-major
                       const float* __restrict__ W2,   // (64,64)
                       const float* __restrict__ W3,   // (64,3)
                       float* __restrict__ out,        // (N,3)
                       int n_points)
{
    // Per-wave enc staging only (h1/h2 transposes are register swizzles).
    __shared__ __align__(16) _Float16 encS[WAVES_PER_BLOCK][32][32]; // [wave][pt][feat]

    const int lane = threadIdx.x & 31;
    const int wave = threadIdx.x >> 5;
    const int col  = lane & 15;
    const int hi   = lane >> 4;

    // ------------------------------------------------------------------
    // Register-resident TRANSPOSED weights as A fragments:
    //   layer L: h^T = W^T(MxK) @ act^T(KxN=16 points)
    // A row m = mb*16+col (output feature), element e -> k via a_klocal.
    // One-time build per persistent wave; weights stay hot in L2.
    // ------------------------------------------------------------------
    v16h A1[4];                       // W1^T: 64x32
#pragma unroll
    for (int mb = 0; mb < 4; ++mb) {
        const int m = mb * 16 + col;
#pragma unroll
        for (int e = 0; e < 16; ++e) {
            const int k = a_klocal(e, hi);            // K_total = 32
            A1[mb][e] = (_Float16)W1[k * 64 + m];     // W1^T[m][k] = W1[k][m]
        }
    }
    v16h A2[4][2];                    // W2^T: 64x64, K split 2x32
#pragma unroll
    for (int mb = 0; mb < 4; ++mb) {
        const int m = mb * 16 + col;
#pragma unroll
        for (int kk = 0; kk < 2; ++kk)
#pragma unroll
            for (int e = 0; e < 16; ++e) {
                const int k = kk * 32 + a_klocal(e, hi);
                A2[mb][kk][e] = (_Float16)W2[k * 64 + m];
            }
    }
    v16h A3[2];                       // W3^T: 3x64 zero-padded to 16 rows
#pragma unroll
    for (int kk = 0; kk < 2; ++kk)
#pragma unroll
        for (int e = 0; e < 16; ++e) {
            const int k = kk * 32 + a_klocal(e, hi);
            A3[kk][e] = (col < 3) ? (_Float16)W3[k * 3 + col] : (_Float16)0.0f;
        }

    const long long waveGid = (long long)blockIdx.x * WAVES_PER_BLOCK + wave;
    const long long nWaves  = (long long)gridDim.x * WAVES_PER_BLOCK;
    const long long nIter   = (n_points + 31) / 32;

    for (long long it = waveGid; it < nIter; it += nWaves) {
        const long long base = it * 32;

        // -------- Phase 0: encode 32 points, one per lane, f32 VALU -------
        long long p = base + lane;
        if (p >= n_points) p = n_points - 1;           // clamp (dup compute, no OOB)
        const float* xp = x + p * 19;
        float f[19];
#pragma unroll
        for (int i = 0; i < 19; ++i) f[i] = __builtin_nontemporal_load(xp + i);

        const float vx = f[0] * 2.0f - 1.0f;
        const float vy = f[1] * 2.0f - 1.0f;
        const float vz = f[2] * 2.0f - 1.0f;
        const float xy = vx * vy, yz = vy * vz, xz = vx * vz;
        const float x2 = vx * vx, y2 = vy * vy, z2 = vz * vz;
        float sh[16];
        sh[0]  = 0.28209479177387814f;
        sh[1]  = -0.48860251190291987f * vy;
        sh[2]  =  0.48860251190291987f * vz;
        sh[3]  = -0.48860251190291987f * vx;
        sh[4]  =  1.0925484305920792f * xy;
        sh[5]  = -1.0925484305920792f * yz;
        sh[6]  =  0.94617469575756f * z2 - 0.31539156525252f;
        sh[7]  = -1.0925484305920792f * xz;
        sh[8]  =  0.5462742152960396f * (x2 - y2);
        sh[9]  =  0.5900435899266435f * vy * (-3.0f * x2 + y2);
        sh[10] =  2.890611442640554f * xy * vz;
        sh[11] =  0.4570457994644657f * vy * (1.0f - 5.0f * z2);
        sh[12] =  0.3731763325901154f * vz * (5.0f * z2 - 3.0f);
        sh[13] =  0.4570457994644657f * vx * (1.0f - 5.0f * z2);
        sh[14] =  1.445305721320277f * vz * (x2 - y2);
        sh[15] =  0.5900435899266435f * vx * (x2 - 3.0f * y2);

        _Float16* er = &encS[wave][lane][0];
#pragma unroll
        for (int i = 0; i < 16; ++i) er[i]      = (_Float16)sh[i];
#pragma unroll
        for (int i = 0; i < 16; ++i) er[16 + i] = (_Float16)f[3 + i];
        lds_phase_fence();

        // -------- Per 16-point tile: 3 WMMA layers, all transposed --------
#pragma unroll
        for (int t = 0; t < 2; ++t) {
            // B1 = enc^T (32x16): lane = point t*16+col, element e -> feature
            // hi*16+e -> 16 CONTIGUOUS halfs -> 2x ds_load_b128.
            const _Float16* brow = &encS[wave][t * 16 + col][hi * 16];
            v8h blo = *(const v8h*)(brow);
            v8h bhh = *(const v8h*)(brow + 8);
            v16h b1;
#pragma unroll
            for (int e = 0; e < 8; ++e) { b1[e] = blo[e]; b1[e + 8] = bhh[e]; }

            const v8f zero = {};

            // ---- Layer 1: h1^T = W1^T @ enc^T ----
            v8f acc1[4];
#pragma unroll
            for (int mb = 0; mb < 4; ++mb)
                acc1[mb] = __builtin_amdgcn_wmma_f32_16x16x32_f16(
                    false, A1[mb], false, b1, (short)0, zero, false, false);

            v16h bh1[2];
            transpose_relu_to_bfrags(acc1, hi, bh1);   // ReLU + D->B (registers)

            // ---- Layer 2: h2^T = W2^T @ h1^T (K=64 as 2x32) ----
            v8f acc2[4];
#pragma unroll
            for (int mb = 0; mb < 4; ++mb) {
                acc2[mb] = __builtin_amdgcn_wmma_f32_16x16x32_f16(
                    false, A2[mb][0], false, bh1[0], (short)0, zero, false, false);
                acc2[mb] = __builtin_amdgcn_wmma_f32_16x16x32_f16(
                    false, A2[mb][1], false, bh1[1], (short)0, acc2[mb], false, false);
            }

            v16h bh2[2];
            transpose_relu_to_bfrags(acc2, hi, bh2);

            // ---- Layer 3: out^T = W3^T @ h2^T ----
            v8f acc3 = __builtin_amdgcn_wmma_f32_16x16x32_f16(
                false, A3[0], false, bh2[0], (short)0, zero, false, false);
            acc3 = __builtin_amdgcn_wmma_f32_16x16x32_f16(
                false, A3[1], false, bh2[1], (short)0, acc3, false, false);

            // D = out^T: lane = point, channels 0..2 in VGPRs 0..2 of the
            // hi==0 half -> one contiguous 12B store per point (coalesced).
            if (hi == 0) {
                const long long row = base + t * 16 + col;
                if (row < n_points) {
                    float* po = out + row * 3;
                    __builtin_nontemporal_store(acc3[0], po + 0);
                    __builtin_nontemporal_store(acc3[1], po + 1);
                    __builtin_nontemporal_store(acc3[2], po + 2);
                }
            }
        }
    }
}

extern "C" void kernel_launch(void* const* d_in, const int* in_sizes, int n_in,
                              void* d_out, int out_size, void* d_ws, size_t ws_size,
                              hipStream_t stream) {
    const float* x  = (const float*)d_in[0];
    const float* W1 = (const float*)d_in[1];
    const float* W2 = (const float*)d_in[2];
    const float* W3 = (const float*)d_in[3];
    float* out = (float*)d_out;
    const int n = in_sizes[0] / 19;

    // Persistent-style grid: keep every wave resident so the one-time
    // weight-fragment build (224 vmem loads) amortizes over ~11 iterations,
    // while still providing ~6K waves of latency hiding.
    const long long nIter = (n + 31) / 32;           // 32 points per wave-iteration
    long long maxBlocks = (nIter + WAVES_PER_BLOCK - 1) / WAVES_PER_BLOCK;
    int blocks = (int)(maxBlocks < 768 ? maxBlocks : 768);
    if (blocks < 1) blocks = 1;

    fused_color_mlp_kernel<<<blocks, 32 * WAVES_PER_BLOCK, 0, stream>>>(
        x, W1, W2, W3, out, n);
}
